// MuliHeadedMaskedSelfAttention_59657095741630
// MI455X (gfx1250) — matrix operand order repaired and checked by reading, not compiled, not run on hardware
//
#include <hip/hip_runtime.h>

typedef __attribute__((ext_vector_type(4)))  int    v4i;
typedef __attribute__((ext_vector_type(8)))  __bf16 v8bf;
typedef __attribute__((ext_vector_type(16))) __bf16 v16bf;
typedef __attribute__((ext_vector_type(8)))  float  v8f;

#define NEG_INF (-__builtin_inff())

// ---- async-to-LDS path (gfx1250 ASYNCcnt), guarded so fallback always compiles ----
#if defined(__has_builtin)
#if __has_builtin(__builtin_amdgcn_global_load_async_to_lds_b128) && \
    __has_builtin(__builtin_amdgcn_s_wait_asynccnt)
#define HAVE_ASYNC_LDS 1
#endif
#endif

static __device__ __forceinline__ void cp16_to_lds(const __bf16* g, __bf16* l) {
#if defined(HAVE_ASYNC_LDS)
  // signature (probe-confirmed): (v4i addrspace(1)* src, v4i addrspace(3)* dst, imm off, imm cpol)
  __builtin_amdgcn_global_load_async_to_lds_b128(
      (__attribute__((address_space(1))) v4i*)g,
      (__attribute__((address_space(3))) v4i*)l, 0, 0);
#else
  *(v8bf*)l = *(const v8bf*)g;
#endif
}

static __device__ __forceinline__ v16bf bf_cat(v8bf lo, v8bf hi) {
  return __builtin_shufflevector(lo, hi, 0,1,2,3,4,5,6,7,8,9,10,11,12,13,14,15);
}
static __device__ __forceinline__ v8f wmma_bf16(v16bf a, v16bf b, v8f c) {
  return __builtin_amdgcn_wmma_f32_16x16x32_bf16(false, a, false, b, (short)0, c, false, false);
}

// ---------------- converters ----------------
__global__ void k_cvt_x(const float* __restrict__ x, __bf16* __restrict__ xb, int n) {
  int i = blockIdx.x * blockDim.x + threadIdx.x;
  if (i < n) xb[i] = (__bf16)x[i];
}

// wqkvt[n][k] = W_{n/1024}[k][n%1024]
__global__ void k_cvt_wqkv(const float* __restrict__ wq, const float* __restrict__ wk,
                           const float* __restrict__ wv, __bf16* __restrict__ wt) {
  int idx = blockIdx.x * blockDim.x + threadIdx.x;
  int nn = idx >> 10, k = idx & 1023;
  int m = nn >> 10, ni = nn & 1023;
  const float* src = (m == 0) ? wq : ((m == 1) ? wk : wv);
  wt[idx] = (__bf16)src[k * 1024 + ni];
}

// wot[z][c] = Wo[c][z]
__global__ void k_cvt_wo(const float* __restrict__ wo, __bf16* __restrict__ wt) {
  int idx = blockIdx.x * blockDim.x + threadIdx.x;
  int z = idx >> 10, c = idx & 1023;
  wt[idx] = (__bf16)wo[c * 1024 + z];
}

// ============ register-blocked, LDS-staged GEMM kernels ============
// Block = 128 threads = 4 waves. Block tile: M=128 (wave w owns 2 m-tiles), N=64 (4 n-tiles,
// shared across waves via a double-buffered LDS B slab of 64 rows x 32 k, rows padded to 40).
#define BSTRIDE 40
#define SLAB    (64 * BSTRIDE)

// ---------------- QKV projection: [4096 x 1024] x [1024 x 3072] ----------------
__global__ __launch_bounds__(128) void k_gemm_qkv(const __bf16* __restrict__ xb,
                                                  const __bf16* __restrict__ wt,
                                                  __bf16* __restrict__ qb,
                                                  __bf16* __restrict__ kb,
                                                  __bf16* __restrict__ vtb) {
  __shared__ __align__(16) __bf16 ldsB[2 * SLAB];
  const int tid = threadIdx.x;
  const int lane = tid & 31, wv = tid >> 5;
  const int nq = lane & 15, hi = lane >> 4;
  const int m0 = blockIdx.x * 128 + wv * 32;   // wave's 32 rows
  const int n0 = blockIdx.y * 64;              // block's 64 cols (uniform slab)
  const __bf16* arow0 = xb + (size_t)(m0 + nq) * 1024;
  const __bf16* arow1 = xb + (size_t)(m0 + 16 + nq) * 1024;
  // staging: 256 chunks of 8 bf16 per k-step; thread copies chunks tid and tid+128
  const int r0 = tid >> 2,         q0 = tid & 3;
  const int r1 = (tid + 128) >> 2, q1 = tid & 3;
  const __bf16* bsrc0 = wt + (size_t)(n0 + r0) * 1024 + q0 * 8;
  const __bf16* bsrc1 = wt + (size_t)(n0 + r1) * 1024 + q1 * 8;
  __bf16* bdst0 = &ldsB[r0 * BSTRIDE + q0 * 8];
  __bf16* bdst1 = &ldsB[r1 * BSTRIDE + q1 * 8];

  v8f acc[8] = {};
  cp16_to_lds(bsrc0, bdst0);
  cp16_to_lds(bsrc1, bdst1);
  int ib = 0;
  for (int kk = 0; kk < 1024; kk += 32, ib ^= 1) {
    const bool more = (kk + 32) < 1024;
    if (more) {
      cp16_to_lds(bsrc0 + kk + 32, bdst0 + (ib ^ 1) * SLAB);
      cp16_to_lds(bsrc1 + kk + 32, bdst1 + (ib ^ 1) * SLAB);
    }
#if defined(HAVE_ASYNC_LDS)
    if (more) __builtin_amdgcn_s_wait_asynccnt(2);
    else      __builtin_amdgcn_s_wait_asynccnt(0);
#endif
    __syncthreads();
    v16bf a0 = bf_cat(*(const v8bf*)(arow0 + kk + hi * 8),
                      *(const v8bf*)(arow0 + kk + 16 + hi * 8));
    v16bf a1 = bf_cat(*(const v8bf*)(arow1 + kk + hi * 8),
                      *(const v8bf*)(arow1 + kk + 16 + hi * 8));
    const __bf16* bbuf = &ldsB[ib * SLAB];
#pragma unroll
    for (int nt = 0; nt < 4; ++nt) {
      const __bf16* bl = bbuf + (nt * 16 + nq) * BSTRIDE + hi * 8;
      v16bf b = bf_cat(*(const v8bf*)(bl), *(const v8bf*)(bl + 16));
      acc[nt]     = wmma_bf16(a0, b, acc[nt]);
      acc[4 + nt] = wmma_bf16(a1, b, acc[4 + nt]);
    }
    __syncthreads();
  }
  const int mi = n0 >> 10;          // uniform: q/k/v select
  const int h = (n0 & 1023) >> 6;   // uniform head (64-wide slab = exactly one head)
#pragma unroll
  for (int mt2 = 0; mt2 < 2; ++mt2) {
#pragma unroll
    for (int nt = 0; nt < 4; ++nt) {
      v8f a = acc[mt2 * 4 + nt];
      const int d = nt * 16 + nq;
#pragma unroll
      for (int r = 0; r < 8; ++r) {
        const int row = m0 + mt2 * 16 + r + 8 * hi;
        const int bb = row >> 11, t = row & 2047;
        const size_t bh = (size_t)(bb * 16 + h);
        const __bf16 val = (__bf16)a[r];
        if (mi == 0)      qb[(bh * 2048 + t) * 64 + d] = val;
        else if (mi == 1) kb[(bh * 2048 + t) * 64 + d] = val;
        else              vtb[(bh * 64 + d) * 2048 + t] = val;  // V transposed [d][t]
      }
    }
  }
}

// ---------------- output projection: [4096 x 1024] x [1024 x 1024] -> f32 ----------------
__global__ __launch_bounds__(128) void k_gemm_out(const __bf16* __restrict__ ao,
                                                  const __bf16* __restrict__ wot,
                                                  float* __restrict__ out) {
  __shared__ __align__(16) __bf16 ldsB[2 * SLAB];
  const int tid = threadIdx.x;
  const int lane = tid & 31, wv = tid >> 5;
  const int nq = lane & 15, hi = lane >> 4;
  const int m0 = blockIdx.x * 128 + wv * 32;
  const int n0 = blockIdx.y * 64;
  const __bf16* arow0 = ao + (size_t)(m0 + nq) * 1024;
  const __bf16* arow1 = ao + (size_t)(m0 + 16 + nq) * 1024;
  const int r0 = tid >> 2,         q0 = tid & 3;
  const int r1 = (tid + 128) >> 2, q1 = tid & 3;
  const __bf16* bsrc0 = wot + (size_t)(n0 + r0) * 1024 + q0 * 8;
  const __bf16* bsrc1 = wot + (size_t)(n0 + r1) * 1024 + q1 * 8;
  __bf16* bdst0 = &ldsB[r0 * BSTRIDE + q0 * 8];
  __bf16* bdst1 = &ldsB[r1 * BSTRIDE + q1 * 8];

  v8f acc[8] = {};
  cp16_to_lds(bsrc0, bdst0);
  cp16_to_lds(bsrc1, bdst1);
  int ib = 0;
  for (int kk = 0; kk < 1024; kk += 32, ib ^= 1) {
    const bool more = (kk + 32) < 1024;
    if (more) {
      cp16_to_lds(bsrc0 + kk + 32, bdst0 + (ib ^ 1) * SLAB);
      cp16_to_lds(bsrc1 + kk + 32, bdst1 + (ib ^ 1) * SLAB);
    }
#if defined(HAVE_ASYNC_LDS)
    if (more) __builtin_amdgcn_s_wait_asynccnt(2);
    else      __builtin_amdgcn_s_wait_asynccnt(0);
#endif
    __syncthreads();
    v16bf a0 = bf_cat(*(const v8bf*)(arow0 + kk + hi * 8),
                      *(const v8bf*)(arow0 + kk + 16 + hi * 8));
    v16bf a1 = bf_cat(*(const v8bf*)(arow1 + kk + hi * 8),
                      *(const v8bf*)(arow1 + kk + 16 + hi * 8));
    const __bf16* bbuf = &ldsB[ib * SLAB];
#pragma unroll
    for (int nt = 0; nt < 4; ++nt) {
      const __bf16* bl = bbuf + (nt * 16 + nq) * BSTRIDE + hi * 8;
      v16bf b = bf_cat(*(const v8bf*)(bl), *(const v8bf*)(bl + 16));
      acc[nt]     = wmma_bf16(a0, b, acc[nt]);
      acc[4 + nt] = wmma_bf16(a1, b, acc[4 + nt]);
    }
    __syncthreads();
  }
#pragma unroll
  for (int mt2 = 0; mt2 < 2; ++mt2) {
#pragma unroll
    for (int nt = 0; nt < 4; ++nt) {
      v8f a = acc[mt2 * 4 + nt];
#pragma unroll
      for (int r = 0; r < 8; ++r) {
        const int row = m0 + mt2 * 16 + r + 8 * hi;
        out[(size_t)row * 1024 + n0 + nt * 16 + nq] = a[r];
      }
    }
  }
}

// ---------------- flash attention: one wave per (b, h, 16-query tile) ----------------
__global__ __launch_bounds__(32) void k_attn(const __bf16* __restrict__ qb,
                                             const __bf16* __restrict__ kb,
                                             const __bf16* __restrict__ vtb,
                                             __bf16* __restrict__ ao) {
  __shared__ __align__(32) __bf16 lds_p[16 * 32];
  const int lane = threadIdx.x & 31;
  const int nq = lane & 15, hi = lane >> 4;
  const int tt = blockIdx.x, h = blockIdx.y, b = blockIdx.z;
  const size_t bh = (size_t)(b * 16 + h);
  const __bf16* qrow = qb + (bh * 2048 + tt * 16 + nq) * 64;
  const __bf16* kp = kb + bh * 2048 * 64;
  const __bf16* vp = vtb + bh * 64 * 2048;

  v16bf aq0 = bf_cat(*(const v8bf*)(qrow + hi * 16),      *(const v8bf*)(qrow + hi * 16 + 8));
  v16bf aq1 = bf_cat(*(const v8bf*)(qrow + 32 + hi * 16), *(const v8bf*)(qrow + 32 + hi * 16 + 8));

  v8f o0 = {}, o1 = {}, o2 = {}, o3 = {};
  float mrun[8], lrun[8];
#pragma unroll
  for (int r = 0; r < 8; ++r) { mrun[r] = NEG_INF; lrun[r] = 0.f; }
  const int tq0 = tt << 4;

  for (int jj = 0; jj <= tq0; jj += 32) {
    const bool v1 = (jj + 16) <= tq0;
    const __bf16* kr0 = kp + (size_t)(jj + nq) * 64;
    v16bf b00 = bf_cat(*(const v8bf*)(kr0 + hi * 16),      *(const v8bf*)(kr0 + hi * 16 + 8));
    v16bf b01 = bf_cat(*(const v8bf*)(kr0 + 32 + hi * 16), *(const v8bf*)(kr0 + 32 + hi * 16 + 8));
    v8f s0 = {};
    s0 = wmma_bf16(aq0, b00, s0);
    s0 = wmma_bf16(aq1, b01, s0);
    v8f s1 = {};
    if (v1) {
      const __bf16* kr1 = kp + (size_t)(jj + 16 + nq) * 64;
      v16bf b10 = bf_cat(*(const v8bf*)(kr1 + hi * 16),      *(const v8bf*)(kr1 + hi * 16 + 8));
      v16bf b11 = bf_cat(*(const v8bf*)(kr1 + 32 + hi * 16), *(const v8bf*)(kr1 + 32 + hi * 16 + 8));
      s1 = wmma_bf16(aq0, b10, s1);
      s1 = wmma_bf16(aq1, b11, s1);
    }
    float p0[8], p1[8];
#pragma unroll
    for (int r = 0; r < 8; ++r) {
      const int tq = tq0 + r + 8 * hi;
      const float c0 = (jj + nq <= tq) ? s0[r] * 8.0f : NEG_INF;
      const float c1 = (v1 && (jj + 16 + nq <= tq)) ? s1[r] * 8.0f : NEG_INF;
      float mx = fmaxf(c0, c1);
      mx = fmaxf(mx, __shfl_xor(mx, 1));
      mx = fmaxf(mx, __shfl_xor(mx, 2));
      mx = fmaxf(mx, __shfl_xor(mx, 4));
      mx = fmaxf(mx, __shfl_xor(mx, 8));
      const float mnew = fmaxf(mrun[r], mx);
      const float e0 = __expf(c0 - mnew), e1 = __expf(c1 - mnew);
      float rs = e0 + e1;
      rs += __shfl_xor(rs, 1); rs += __shfl_xor(rs, 2);
      rs += __shfl_xor(rs, 4); rs += __shfl_xor(rs, 8);
      const float corr = __expf(mrun[r] - mnew);
      lrun[r] = lrun[r] * corr + rs;
      mrun[r] = mnew;
      o0[r] = o0[r] * corr; o1[r] = o1[r] * corr;
      o2[r] = o2[r] * corr; o3[r] = o3[r] * corr;
      p0[r] = e0; p1[r] = e1;
    }
    __syncthreads();
#pragma unroll
    for (int r = 0; r < 8; ++r) {
      lds_p[(r + 8 * hi) * 32 + nq]      = (__bf16)p0[r];
      lds_p[(r + 8 * hi) * 32 + 16 + nq] = (__bf16)p1[r];
    }
    __syncthreads();
    const __bf16* prow = lds_p + nq * 32;
    v16bf ap = bf_cat(*(const v8bf*)(prow + hi * 16), *(const v8bf*)(prow + hi * 16 + 8));
    const int cko = (hi && v1) ? 16 : 0;
    {
      const __bf16* vr = vp + (size_t)(0 * 16 + nq) * 2048 + jj + cko;
      o0 = wmma_bf16(ap, bf_cat(*(const v8bf*)(vr), *(const v8bf*)(vr + 8)), o0);
    }
    {
      const __bf16* vr = vp + (size_t)(1 * 16 + nq) * 2048 + jj + cko;
      o1 = wmma_bf16(ap, bf_cat(*(const v8bf*)(vr), *(const v8bf*)(vr + 8)), o1);
    }
    {
      const __bf16* vr = vp + (size_t)(2 * 16 + nq) * 2048 + jj + cko;
      o2 = wmma_bf16(ap, bf_cat(*(const v8bf*)(vr), *(const v8bf*)(vr + 8)), o2);
    }
    {
      const __bf16* vr = vp + (size_t)(3 * 16 + nq) * 2048 + jj + cko;
      o3 = wmma_bf16(ap, bf_cat(*(const v8bf*)(vr), *(const v8bf*)(vr + 8)), o3);
    }
  }
#pragma unroll
  for (int r = 0; r < 8; ++r) {
    const float inv = 1.0f / lrun[r];
    const int t = tq0 + r + 8 * hi;
    __bf16* orow = ao + ((size_t)b * 2048 + t) * 1024 + h * 64;
    orow[0 + nq]  = (__bf16)(o0[r] * inv);
    orow[16 + nq] = (__bf16)(o1[r] * inv);
    orow[32 + nq] = (__bf16)(o2[r] * inv);
    orow[48 + nq] = (__bf16)(o3[r] * inv);
  }
}

extern "C" void kernel_launch(void* const* d_in, const int* in_sizes, int n_in,
                              void* d_out, int out_size, void* d_ws, size_t ws_size,
                              hipStream_t stream) {
  (void)in_sizes; (void)n_in; (void)out_size; (void)ws_size;
  const float* x  = (const float*)d_in[0];
  const float* wq = (const float*)d_in[1];
  const float* wk = (const float*)d_in[2];
  const float* wv = (const float*)d_in[3];
  const float* wo = (const float*)d_in[4];

  char* w = (char*)d_ws;                       // ~48 MB workspace layout
  __bf16* xb    = (__bf16*)(w);                // [4096][1024]
  __bf16* wqkvt = (__bf16*)(w + (8u  << 20));  // [3072][1024]
  __bf16* wot   = (__bf16*)(w + (14u << 20));  // [1024][1024]
  __bf16* qb    = (__bf16*)(w + (16u << 20));  // [B][H][T][DK]
  __bf16* kb    = (__bf16*)(w + (24u << 20));  // [B][H][T][DK]
  __bf16* vtb   = (__bf16*)(w + (32u << 20));  // [B][H][DK][T]
  __bf16* ao    = (__bf16*)(w + (40u << 20));  // [4096][1024]

  k_cvt_x   <<<(4096 * 1024) / 256, 256, 0, stream>>>(x, xb, 4096 * 1024);
  k_cvt_wqkv<<<(3072 * 1024) / 256, 256, 0, stream>>>(wq, wk, wv, wqkvt);
  k_cvt_wo  <<<(1024 * 1024) / 256, 256, 0, stream>>>(wo, wot);
  k_gemm_qkv<<<dim3(32, 48), 128, 0, stream>>>(xb, wqkvt, qb, kb, vtb);
  k_attn    <<<dim3(128, 16, 2), 32, 0, stream>>>(qb, kb, vtb, ao);
  k_gemm_out<<<dim3(32, 16), 128, 0, stream>>>(ao, wot, (float*)d_out);
}
